// RowWiseTransformers_26087631356375
// MI455X (gfx1250) — compile-verified, matrix operand order checked by reading, not run on hardware
//
#include <hip/hip_runtime.h>
#include <math.h>

typedef __attribute__((ext_vector_type(16))) _Float16 v16h;
typedef __attribute__((ext_vector_type(8)))  float    v8f;
typedef unsigned int v4u __attribute__((ext_vector_type(4)));
typedef int          v8i __attribute__((ext_vector_type(8)));
typedef int          v4i __attribute__((ext_vector_type(4)));

#define LAYERS 6
#define DMODEL 512
#define NHEAD  8
#define SEQF   256
#define DMLP   2048
#define BATCH  256
#define HDIM   64
#define NROWS  (BATCH * SEQF)

// ---------------------------------------------------------------------------
// WMMA fragment helpers (CDNA5 16-bit A/B layout, wave32):
//   lane L: m(or n) = L & 15 ; K-half offset = (L>>4)*8
//   VGPR p (0..7): K = (p<4 ? 0 : 16) + (p&3)*2 (+ khalf), two halfs per VGPR.
// Pairs (j, j+1) are adjacent in K -> contiguous b32/b128 loads.
// ---------------------------------------------------------------------------
__device__ inline v16h load_frag(const _Float16* __restrict__ base, int ld,
                                 int r0, int k0) {
  int lane  = threadIdx.x & 31;
  int m     = lane & 15;
  int khalf = (lane >> 4) << 3;
  const _Float16* p = base + (size_t)(r0 + m) * (size_t)ld + k0 + khalf;
  v16h f;
#pragma unroll
  for (int pp = 0; pp < 8; ++pp) {
    int k = ((pp & 4) << 2) + ((pp & 3) << 1);  // 0..7 & 16..23 pattern
    f[2 * pp]     = p[k];
    f[2 * pp + 1] = p[k + 1];
  }
  return f;
}

// B fragment straight from a row-major [K][N] matrix (strided in K).
__device__ inline v16h load_frag_strided(const _Float16* __restrict__ base,
                                         int ld, int n0, int k0) {
  int lane  = threadIdx.x & 31;
  int n     = lane & 15;
  int khalf = (lane >> 4) << 3;
  v16h f;
#pragma unroll
  for (int j = 0; j < 16; ++j) {
    int pp = j >> 1;
    int k  = khalf + ((pp & 4) << 2) + ((pp & 3) << 1) + (j & 1);
    f[j] = base[(size_t)(k0 + k) * (size_t)ld + n0 + n];
  }
  return f;
}

__device__ inline v8f wmma16(v16h a, v16h b, v8f c) {
  return __builtin_amdgcn_wmma_f32_16x16x32_f16(false, a, false, b, (short)0, c,
                                                false, false);
}

// ---------------------------------------------------------------------------
// Tensor Data Mover: DMA a 128x32 f16 tile (rows x K-slice) of the weight
// matrix Bt[N][K] into LDS.  D# per CDNA5 ISA:
//   group0: count=1 | lds_addr | global_addr(57b) | type=2
//   group1: data_size=1(2B) | tensor_dim0=K | tensor_dim1=128 |
//           tile_dim0=32 | tile_dim1=128 | tensor_dim0_stride=K
//   groups 2/3: zero (2-D tensor).
// ---------------------------------------------------------------------------
__device__ inline void tdm_fetch_b(_Float16* ldsDst, const _Float16* gsrc,
                                   int ldK) {
  unsigned lds_addr = (unsigned)(size_t)ldsDst;  // addr[31:0] = LDS offset
  unsigned long long ga = (unsigned long long)(size_t)gsrc;
  v4u g0;
  g0[0] = 1u;                                        // count=1, user D#
  g0[1] = lds_addr;
  g0[2] = (unsigned)(ga & 0xffffffffu);
  g0[3] = (unsigned)((ga >> 32) & 0x01ffffffu) | (2u << 30);  // type=2
  unsigned K = (unsigned)ldK;
  v8i g1;
  g1[0] = (int)(1u << 16);                    // data_size=1 -> 2 bytes
  g1[1] = (int)((K & 0xffffu) << 16);         // tensor_dim0[15:0]
  g1[2] = (int)(((K >> 16) & 0xffffu) | (128u << 16));  // dim0 hi | dim1 lo
  g1[3] = (int)(32u << 16);                   // dim1 hi=0 | tile_dim0=32
  g1[4] = 128;                                // tile_dim1=128, tile_dim2=0
  g1[5] = (int)K;                             // tensor_dim0_stride[31:0]
  g1[6] = 0;                                  // stride hi | dim1_stride lo
  g1[7] = 0;
  v4i z4 = {0, 0, 0, 0};
#if defined(__clang_major__) && (__clang_major__ >= 23)
  v8i z8 = {0, 0, 0, 0, 0, 0, 0, 0};
  __builtin_amdgcn_tensor_load_to_lds(g0, g1, z4, z4, z8, 0);
#else
  __builtin_amdgcn_tensor_load_to_lds(g0, g1, z4, z4, 0);
#endif
}

// ---------------------------------------------------------------------------
// WMMA GEMM: C = A[M,K](f16) * Bt[N,K](f16)^T + bias
// MODE 0: store f16   MODE 1: f32 residual add (h += ...)   MODE 2: GELU->f16
// Block: 128 thr = 4 waves (2 row x 2 col), wave tile 32x64, block tile 64x128.
// B tile (128 cols x 32 K) staged into LDS by the TDM, double buffered.
// ---------------------------------------------------------------------------
template <int MODE>
__global__ __launch_bounds__(128) void gemm_kernel(
    const _Float16* __restrict__ A, const _Float16* __restrict__ Bt,
    const float* __restrict__ bias, float* __restrict__ Cf,
    _Float16* __restrict__ Ch, int M, int N, int K) {
  __shared__ _Float16 Btile[2][128 * 32];  // 2 x 8 KB
  int wave = threadIdx.x >> 5;
  int lane = threadIdx.x & 31;
  int r0   = blockIdx.x * 64 + (wave >> 1) * 32;
  int cblk = blockIdx.y * 128;
  int cw   = (wave & 1) * 64;  // wave's column offset inside the LDS tile
  int nk   = K >> 5;

  v8f acc[2][4];
#pragma unroll
  for (int mi = 0; mi < 2; ++mi)
#pragma unroll
    for (int ni = 0; ni < 4; ++ni)
#pragma unroll
      for (int e = 0; e < 8; ++e) acc[mi][ni][e] = 0.0f;

  const _Float16* Bblk = Bt + (size_t)cblk * (size_t)K;
  if (wave == 0) tdm_fetch_b(&Btile[0][0], Bblk, K);

  for (int i = 0; i < nk; ++i) {
    if (wave == 0) {
      if (i + 1 < nk) {
        tdm_fetch_b(&Btile[(i + 1) & 1][0], Bblk + (size_t)(i + 1) * 32, K);
        __builtin_amdgcn_s_wait_tensorcnt(1);  // oldest (current buf) done
      } else {
        __builtin_amdgcn_s_wait_tensorcnt(0);
      }
    }
    __syncthreads();  // tile i visible to all waves
    int k0 = i << 5;
    v16h a0 = load_frag(A, K, r0, k0);
    v16h a1 = load_frag(A, K, r0 + 16, k0);
    const _Float16* bl = &Btile[i & 1][0];
#pragma unroll
    for (int ni = 0; ni < 4; ++ni) {
      v16h bf = load_frag(bl, 32, cw + ni * 16, 0);
      acc[0][ni] = wmma16(a0, bf, acc[0][ni]);
      acc[1][ni] = wmma16(a1, bf, acc[1][ni]);
    }
    __syncthreads();  // readers done before TDM overwrites this buffer
  }

  int n = lane & 15, half = lane >> 4;
#pragma unroll
  for (int mi = 0; mi < 2; ++mi)
#pragma unroll
    for (int ni = 0; ni < 4; ++ni) {
      int col = cblk + cw + ni * 16 + n;
      float bv = bias[col];
#pragma unroll
      for (int r = 0; r < 8; ++r) {
        int row = r0 + mi * 16 + r + 8 * half;
        float val = acc[mi][ni][r] + bv;
        if (MODE == 0) {
          Ch[(size_t)row * N + col] = (_Float16)val;
        } else if (MODE == 1) {
          Cf[(size_t)row * N + col] += val;
        } else {
          float g = 0.5f * val * (1.0f + erff(val * 0.70710678118f));
          Ch[(size_t)row * N + col] = (_Float16)g;
        }
      }
    }
}

// ---------------------------------------------------------------------------
// Flash attention: grid (B*H, F/64), 128 thr = 4 waves; wave owns 16 q-rows.
// ---------------------------------------------------------------------------
__global__ __launch_bounds__(128) void attn_kernel(
    const _Float16* __restrict__ qp, const _Float16* __restrict__ kp,
    const _Float16* __restrict__ vp, _Float16* __restrict__ op) {
  __shared__ _Float16 Pld[4][16 * 32];
  int wave = threadIdx.x >> 5;
  int lane = threadIdx.x & 31;
  int b  = blockIdx.x / NHEAD;
  int hh = blockIdx.x % NHEAD;
  int qrow = blockIdx.y * 64 + wave * 16;

  const _Float16* Q  = qp + (size_t)b * SEQF * DMODEL + hh * HDIM;
  const _Float16* Kb = kp + (size_t)b * SEQF * DMODEL + hh * HDIM;
  const _Float16* Vb = vp + (size_t)b * SEQF * DMODEL + hh * HDIM;
  _Float16*       Ob = op + (size_t)b * SEQF * DMODEL + hh * HDIM;

  v16h qa0 = load_frag(Q, DMODEL, qrow, 0);
  v16h qa1 = load_frag(Q, DMODEL, qrow, 32);

  v8f oacc[4];
  float mrun[8], lrun[8];
#pragma unroll
  for (int j = 0; j < 4; ++j)
#pragma unroll
    for (int e = 0; e < 8; ++e) oacc[j][e] = 0.0f;
#pragma unroll
  for (int r = 0; r < 8; ++r) { mrun[r] = -3.0e38f; lrun[r] = 0.0f; }

  _Float16* myP = &Pld[wave][0];
  int n = lane & 15, half = lane >> 4;

  for (int kb = 0; kb < SEQF; kb += 32) {
    // ---- S = Q K^T / sqrt(HD), two 16-wide key tiles per step ----
    v8f s[2];
#pragma unroll
    for (int t = 0; t < 2; ++t) {
#pragma unroll
      for (int e = 0; e < 8; ++e) s[t][e] = 0.0f;
      v16h kb0 = load_frag(Kb, DMODEL, kb + t * 16, 0);
      v16h kb1 = load_frag(Kb, DMODEL, kb + t * 16, 32);
      s[t] = wmma16(qa0, kb0, s[t]);
      s[t] = wmma16(qa1, kb1, s[t]);
    }
    // ---- online softmax row stats (rows r+8*half, cols across lanes) ----
    float mx[8];
#pragma unroll
    for (int r = 0; r < 8; ++r) {
      s[0][r] *= 0.125f;  // 1/sqrt(64)
      s[1][r] *= 0.125f;
      mx[r] = fmaxf(s[0][r], s[1][r]);
    }
#pragma unroll
    for (int off = 1; off < 16; off <<= 1)
#pragma unroll
      for (int r = 0; r < 8; ++r)
        mx[r] = fmaxf(mx[r], __shfl_xor(mx[r], off, 32));

    float fac[8], psum[8];
#pragma unroll
    for (int r = 0; r < 8; ++r) {
      float mnew = fmaxf(mrun[r], mx[r]);
      fac[r]  = __expf(mrun[r] - mnew);
      s[0][r] = __expf(s[0][r] - mnew);
      s[1][r] = __expf(s[1][r] - mnew);
      psum[r] = s[0][r] + s[1][r];
      mrun[r] = mnew;
    }
#pragma unroll
    for (int off = 1; off < 16; off <<= 1)
#pragma unroll
      for (int r = 0; r < 8; ++r) psum[r] += __shfl_xor(psum[r], off, 32);
#pragma unroll
    for (int r = 0; r < 8; ++r) {
      lrun[r] = lrun[r] * fac[r] + psum[r];
#pragma unroll
      for (int j = 0; j < 4; ++j) oacc[j][r] *= fac[r];
    }
    // ---- stage P through LDS: C-layout -> row-major 16x32 -> A-layout ----
#pragma unroll
    for (int r = 0; r < 8; ++r) {
      myP[(r + 8 * half) * 32 + n]      = (_Float16)s[0][r];
      myP[(r + 8 * half) * 32 + 16 + n] = (_Float16)s[1][r];
    }
    asm volatile("s_wait_dscnt 0" ::: "memory");
    v16h pf = load_frag(myP, 32, 0, 0);
    // ---- O += P @ V over this 32-key block ----
#pragma unroll
    for (int j = 0; j < 4; ++j) {
      v16h vb = load_frag_strided(Vb, DMODEL, j * 16, kb);
      oacc[j] = wmma16(pf, vb, oacc[j]);
    }
  }
  // ---- normalize + store ----
#pragma unroll
  for (int r = 0; r < 8; ++r) {
    float inv = 1.0f / lrun[r];
    int row = qrow + r + 8 * half;
#pragma unroll
    for (int j = 0; j < 4; ++j)
      Ob[(size_t)row * DMODEL + j * 16 + n] = (_Float16)(oacc[j][r] * inv);
  }
}

// ---------------------------------------------------------------------------
// LayerNorm over D=512, one block per row, output f16 (GEMM operand).
// ---------------------------------------------------------------------------
__global__ __launch_bounds__(256) void ln_kernel(const float* __restrict__ h,
                                                 const float* __restrict__ g,
                                                 const float* __restrict__ bta,
                                                 _Float16* __restrict__ outp) {
  __shared__ float red[256];
  size_t row = blockIdx.x;
  int tid = threadIdx.x;
  const float* hr = h + row * DMODEL;
  float x0 = hr[tid], x1 = hr[tid + 256];
  red[tid] = x0 + x1;
  __syncthreads();
  for (int off = 128; off > 0; off >>= 1) {
    if (tid < off) red[tid] += red[tid + off];
    __syncthreads();
  }
  float mean = red[0] * (1.0f / DMODEL);
  __syncthreads();
  float d0 = x0 - mean, d1 = x1 - mean;
  red[tid] = d0 * d0 + d1 * d1;
  __syncthreads();
  for (int off = 128; off > 0; off >>= 1) {
    if (tid < off) red[tid] += red[tid + off];
    __syncthreads();
  }
  float rstd = rsqrtf(red[0] * (1.0f / DMODEL) + 1e-5f);
  outp[row * DMODEL + tid]       = (_Float16)(d0 * rstd * g[tid] + bta[tid]);
  outp[row * DMODEL + tid + 256] = (_Float16)(d1 * rstd * g[tid + 256] + bta[tid + 256]);
}

// ---------------------------------------------------------------------------
// Embedding: clip, row-normalize x (ddof=1), h = xn*ew + eb + pos. Block per b.
// ---------------------------------------------------------------------------
__global__ __launch_bounds__(256) void embed_kernel(
    const float* __restrict__ x, const float* __restrict__ ew,
    const float* __restrict__ eb, const float* __restrict__ pe,
    float* __restrict__ h) {
  __shared__ float red[256];
  __shared__ float xn[SEQF];
  int b = blockIdx.x, tid = threadIdx.x;
  float xv = x[(size_t)b * SEQF + tid];
  xv = fminf(fmaxf(xv, -10000.0f), 10000.0f);
  red[tid] = xv;
  __syncthreads();
  for (int off = 128; off > 0; off >>= 1) {
    if (tid < off) red[tid] += red[tid + off];
    __syncthreads();
  }
  float mu = red[0] * (1.0f / SEQF);
  __syncthreads();
  float d = xv - mu;
  red[tid] = d * d;
  __syncthreads();
  for (int off = 128; off > 0; off >>= 1) {
    if (tid < off) red[tid] += red[tid + off];
    __syncthreads();
  }
  float sd = sqrtf(red[0] * (1.0f / (SEQF - 1))) + 1e-8f;
  xn[tid] = d / sd;
  __syncthreads();
  size_t base = (size_t)b * SEQF * DMODEL;
  for (int idx = tid; idx < SEQF * DMODEL; idx += 256) {
    int f = idx >> 9, dd = idx & (DMODEL - 1);
    h[base + idx] = xn[f] * ew[dd] + eb[dd] + pe[idx];
  }
}

// ---------------------------------------------------------------------------
// Weight prep: W[K][N] f32 -> Wt[N][K] f16 (B-operand layout, contiguous K).
// ---------------------------------------------------------------------------
__global__ void wprep_kernel(const float* __restrict__ W,
                             _Float16* __restrict__ Wt, int K, int N) {
  int idx = blockIdx.x * 256 + threadIdx.x;
  if (idx < K * N) {
    int nn = idx / K, kk = idx % K;
    Wt[idx] = (_Float16)W[(size_t)kk * N + nn];
  }
}

// ---------------------------------------------------------------------------
// Pooling + output head. One block per batch row.
// ---------------------------------------------------------------------------
__global__ __launch_bounds__(256) void pool_kernel(
    const float* __restrict__ h, const float* __restrict__ pw,
    const float* __restrict__ ow, const float* __restrict__ ob,
    float* __restrict__ outp) {
  __shared__ float sc[SEQF];
  __shared__ float red[256];
  int b = blockIdx.x, tid = threadIdx.x;
  const float* hr = h + ((size_t)b * SEQF + tid) * DMODEL;
  float s = 0.0f;
  for (int d = 0; d < DMODEL; ++d) s += hr[d] * pw[d];
  red[tid] = s;
  __syncthreads();
  for (int off = 128; off > 0; off >>= 1) {
    if (tid < off) red[tid] = fmaxf(red[tid], red[tid + off]);
    __syncthreads();
  }
  float mx = red[0];
  __syncthreads();
  float e = __expf(s - mx);
  red[tid] = e;
  __syncthreads();
  for (int off = 128; off > 0; off >>= 1) {
    if (tid < off) red[tid] += red[tid + off];
    __syncthreads();
  }
  sc[tid] = e / red[0];
  __syncthreads();
  float acc = 0.0f;
  const float* hb = h + (size_t)b * SEQF * DMODEL;
  for (int d = tid; d < DMODEL; d += 256) {
    float pd = 0.0f;
    for (int f = 0; f < SEQF; ++f) pd += sc[f] * hb[(size_t)f * DMODEL + d];
    acc += pd * ow[d];
  }
  red[tid] = acc;
  __syncthreads();
  for (int off = 128; off > 0; off >>= 1) {
    if (tid < off) red[tid] += red[tid + off];
    __syncthreads();
  }
  if (tid == 0) outp[b] = red[0] + ob[0];
}

// ---------------------------------------------------------------------------
extern "C" void kernel_launch(void* const* d_in, const int* in_sizes, int n_in,
                              void* d_out, int out_size, void* d_ws,
                              size_t ws_size, hipStream_t stream) {
  const float* x    = (const float*)d_in[0];
  const float* ew   = (const float*)d_in[1];
  const float* eb   = (const float*)d_in[2];
  const float* pe   = (const float*)d_in[3];
  const float* ln1g = (const float*)d_in[4];
  const float* ln1b = (const float*)d_in[5];
  const float* wq   = (const float*)d_in[6];
  const float* bq   = (const float*)d_in[7];
  const float* wk   = (const float*)d_in[8];
  const float* bk   = (const float*)d_in[9];
  const float* wv   = (const float*)d_in[10];
  const float* bv   = (const float*)d_in[11];
  const float* wo   = (const float*)d_in[12];
  const float* bo   = (const float*)d_in[13];
  const float* ln2g = (const float*)d_in[14];
  const float* ln2b = (const float*)d_in[15];
  const float* w1   = (const float*)d_in[16];
  const float* b1   = (const float*)d_in[17];
  const float* w2   = (const float*)d_in[18];
  const float* b2   = (const float*)d_in[19];
  const float* pw   = (const float*)d_in[20];
  const float* ow   = (const float*)d_in[21];
  const float* ob   = (const float*)d_in[22];
  float* outp = (float*)d_out;

  char* ws = (char*)d_ws;
  size_t off = 0;
  auto take = [&](size_t bytes) -> char* {
    char* p = ws + off;
    off += (bytes + 255) & ~(size_t)255;
    return p;
  };
  const size_t nDD = (size_t)DMODEL * DMODEL;
  const size_t nDM = (size_t)DMODEL * DMLP;
  _Float16* wtq = (_Float16*)take(LAYERS * nDD * 2);
  _Float16* wtk = (_Float16*)take(LAYERS * nDD * 2);
  _Float16* wtv = (_Float16*)take(LAYERS * nDD * 2);
  _Float16* wto = (_Float16*)take(LAYERS * nDD * 2);
  _Float16* wt1 = (_Float16*)take(LAYERS * nDM * 2);  // [MLP][D]
  _Float16* wt2 = (_Float16*)take(LAYERS * nDM * 2);  // [D][MLP]
  float*    h   = (float*)take((size_t)NROWS * DMODEL * 4);
  _Float16* a   = (_Float16*)take((size_t)NROWS * DMODEL * 2);
  _Float16* qa  = (_Float16*)take((size_t)NROWS * DMODEL * 2);
  _Float16* ka  = (_Float16*)take((size_t)NROWS * DMODEL * 2);
  _Float16* va  = (_Float16*)take((size_t)NROWS * DMODEL * 2);
  _Float16* oa  = (_Float16*)take((size_t)NROWS * DMODEL * 2);
  _Float16* u   = (_Float16*)take((size_t)NROWS * DMLP * 2);

  // ---- one-time weight conversion/transpose (f32 -> f16 [N][K]) ----
  int gDDn = (int)((nDD + 255) / 256), gDMn = (int)((nDM + 255) / 256);
  for (int l = 0; l < LAYERS; ++l) {
    wprep_kernel<<<gDDn, 256, 0, stream>>>(wq + l * nDD, wtq + l * nDD, DMODEL, DMODEL);
    wprep_kernel<<<gDDn, 256, 0, stream>>>(wk + l * nDD, wtk + l * nDD, DMODEL, DMODEL);
    wprep_kernel<<<gDDn, 256, 0, stream>>>(wv + l * nDD, wtv + l * nDD, DMODEL, DMODEL);
    wprep_kernel<<<gDDn, 256, 0, stream>>>(wo + l * nDD, wto + l * nDD, DMODEL, DMODEL);
    wprep_kernel<<<gDMn, 256, 0, stream>>>(w1 + l * nDM, wt1 + l * nDM, DMODEL, DMLP);
    wprep_kernel<<<gDMn, 256, 0, stream>>>(w2 + l * nDM, wt2 + l * nDM, DMLP, DMODEL);
  }

  embed_kernel<<<BATCH, 256, 0, stream>>>(x, ew, eb, pe, h);

  dim3 gD(NROWS / 64, DMODEL / 128);
  dim3 gM(NROWS / 64, DMLP / 128);
  dim3 gA(BATCH * NHEAD, SEQF / 64);

  for (int l = 0; l < LAYERS; ++l) {
    ln_kernel<<<NROWS, 256, 0, stream>>>(h, ln1g + l * DMODEL, ln1b + l * DMODEL, a);
    gemm_kernel<0><<<gD, 128, 0, stream>>>(a, wtq + l * nDD, bq + l * DMODEL, nullptr, qa, NROWS, DMODEL, DMODEL);
    gemm_kernel<0><<<gD, 128, 0, stream>>>(a, wtk + l * nDD, bk + l * DMODEL, nullptr, ka, NROWS, DMODEL, DMODEL);
    gemm_kernel<0><<<gD, 128, 0, stream>>>(a, wtv + l * nDD, bv + l * DMODEL, nullptr, va, NROWS, DMODEL, DMODEL);
    attn_kernel<<<gA, 128, 0, stream>>>(qa, ka, va, oa);
    gemm_kernel<1><<<gD, 128, 0, stream>>>(oa, wto + l * nDD, bo + l * DMODEL, h, nullptr, NROWS, DMODEL, DMODEL);
    ln_kernel<<<NROWS, 256, 0, stream>>>(h, ln2g + l * DMODEL, ln2b + l * DMODEL, a);
    gemm_kernel<2><<<gM, 128, 0, stream>>>(a, wt1 + l * nDM, b1 + l * DMLP, nullptr, u, NROWS, DMLP, DMODEL);
    gemm_kernel<1><<<gD, 128, 0, stream>>>(u, wt2 + l * nDM, b2 + l * DMODEL, h, nullptr, NROWS, DMODEL, DMLP);
  }

  pool_kernel<<<BATCH, 256, 0, stream>>>(h, pw, ow, ob, outp);
}
